// HammingLoss_70729521431100
// MI455X (gfx1250) — 8-bit matrix operand order repaired and checked by reading, compiled, not run on hardware
//
#include <hip/hip_runtime.h>
#include <hip/hip_bf16.h>

#define NUM_CLASSES 1000
#define CODE_LEN    512
#define PAD_CLASSES 1008
#define NTILES      63

typedef __attribute__((ext_vector_type(8))) int v8i;

__global__ void hl_init_acc(unsigned long long* acc) {
    if (threadIdx.x == 0 && blockIdx.x == 0) acc[0] = 0ULL;
}

__global__ void hl_pack_codebook(const float* __restrict__ cb,
                                 signed char* __restrict__ C8,
                                 int* __restrict__ w) {
    const int row = blockIdx.x;
    const int t   = threadIdx.x;
    int s = 0;
    if (row < NUM_CLASSES) {
        #pragma unroll
        for (int j = t; j < CODE_LEN; j += 128) {
            const float f = cb[row * CODE_LEN + j];
            const int c = (f != 0.0f) ? 1 : 0;
            C8[row * CODE_LEN + j] = (signed char)c;
            s += c;
        }
    } else {
        #pragma unroll
        for (int j = t; j < CODE_LEN; j += 128) {
            C8[row * CODE_LEN + j] = 0;
        }
    }
    __shared__ int red[128];
    red[t] = s;
    __syncthreads();
    for (int off = 64; off > 0; off >>= 1) {
        if (t < off) red[t] += red[t + off];
        __syncthreads();
    }
    if (t == 0) w[row] = red[0];
}

__global__ void __launch_bounds__(32)
hl_lut_wmma(const signed char* __restrict__ C8,
            const int* __restrict__ w,
            unsigned short* __restrict__ lut) {
    const int tileJ = blockIdx.x;
    const int tileI = blockIdx.y;
    const int lane  = threadIdx.x;
    const int L     = lane & 15;
    const int h     = lane >> 4;

    const int rowA = tileI * 16 + L;
    const int rowB = tileJ * 16 + L;

    const int* a32 = reinterpret_cast<const int*>(C8) + rowA * (CODE_LEN / 4);
    const int* b32 = reinterpret_cast<const int*>(C8) + rowB * (CODE_LEN / 4);

    v8i acc = {};

    #pragma unroll
    for (int kk = 0; kk < CODE_LEN; kk += 64) {
        const int ai = (kk >> 2) + 4 * h;
        const int2 a0 = *reinterpret_cast<const int2*>(a32 + ai);
        const int2 a1 = *reinterpret_cast<const int2*>(a32 + ai + 2);
        const int2 a2 = *reinterpret_cast<const int2*>(a32 + ai + 8);
        const int2 a3 = *reinterpret_cast<const int2*>(a32 + ai + 10);
        v8i a = { a0.x, a0.y, a1.x, a1.y, a2.x, a2.y, a3.x, a3.y };

        const int bi = (kk >> 2) + 4 * h;
        const int4 b0 = *reinterpret_cast<const int4*>(b32 + bi);
        const int4 b1 = *reinterpret_cast<const int4*>(b32 + bi + 8);
        v8i b = { b0.x, b0.y, b0.z, b0.w, b1.x, b1.y, b1.z, b1.w };

        acc = __builtin_amdgcn_wmma_i32_16x16x64_iu8(false, a, false, b, acc,
                                                     false, false);
    }

    const int n  = tileJ * 16 + L;
    const int wn = w[n];
    #pragma unroll
    for (int r = 0; r < 8; ++r) {
        const int m = tileI * 16 + r + 8 * h;
        const int d = w[m] + wn - 2 * acc[r];
        lut[m * PAD_CLASSES + n] = (unsigned short)d;
    }
}

__global__ void hl_gather_reduce(const int* __restrict__ outp,
                                 const int* __restrict__ targ,
                                 const unsigned short* __restrict__ lut,
                                 unsigned long long* acc, int n) {
    unsigned int s = 0;
    for (int i = blockIdx.x * blockDim.x + threadIdx.x; i < n;
         i += gridDim.x * blockDim.x) {
        const int p = outp[i];
        const int g = targ[i];
        s += (unsigned int)lut[p * PAD_CLASSES + g];
    }
    __shared__ unsigned int red[256];
    const int t = threadIdx.x;
    red[t] = s;
    __syncthreads();
    for (int off = 128; off > 0; off >>= 1) {
        if (t < off) red[t] += red[t + off];
        __syncthreads();
    }
    if (t == 0) atomicAdd(acc, (unsigned long long)red[0]);
}

__global__ void hl_finalize(const unsigned long long* __restrict__ acc,
                            float* __restrict__ out, int n) {
    if (threadIdx.x == 0 && blockIdx.x == 0) {
        out[0] = (float)((double)acc[0] / (double)n);
    }
}

extern "C" void kernel_launch(void* const* d_in, const int* in_sizes, int n_in,
                              void* d_out, int out_size, void* d_ws, size_t ws_size,
                              hipStream_t stream) {
    const int*   outp = (const int*)d_in[0];
    const int*   targ = (const int*)d_in[1];
    const float* cb   = (const float*)d_in[2];
    const int    B    = in_sizes[0];
    float* out = (float*)d_out;

    char* ws = (char*)d_ws;
    signed char*        C8  = (signed char*)ws;
    int*                w   = (int*)(ws + 516096);
    unsigned short*     lut = (unsigned short*)(ws + 520128);
    unsigned long long* acc = (unsigned long long*)(ws + 2552256);

    hl_init_acc<<<1, 1, 0, stream>>>(acc);
    hl_pack_codebook<<<PAD_CLASSES, 128, 0, stream>>>(cb, C8, w);
    hl_lut_wmma<<<dim3(NTILES, NTILES), 32, 0, stream>>>(C8, w, lut);
    hl_gather_reduce<<<512, 256, 0, stream>>>(outp, targ, lut, acc, B);
    hl_finalize<<<1, 1, 0, stream>>>(acc, out, B);
}
